// GeneratorModel_36378372997512
// MI455X (gfx1250) — matrix operand order repaired and checked by reading, not compiled, not run on hardware
//
#include <hip/hip_runtime.h>
#include <hip/hip_bf16.h>
#include <math.h>
#include <stdint.h>

// ---------------------------------------------------------------------------
// Types for CDNA5 WMMA (gfx1250): v16bf A/B fragments, v8f accumulator.
// ---------------------------------------------------------------------------
typedef __bf16 bf16_t;
typedef __attribute__((ext_vector_type(16))) __bf16 v16bf;
typedef __attribute__((ext_vector_type(8)))  __bf16 v8bf;
typedef __attribute__((ext_vector_type(8)))  float  v8f;

static __device__ __forceinline__ bf16_t f2bf(float f) {
    unsigned u = __builtin_bit_cast(unsigned, f);
    unsigned r = (u + 0x7FFFu + ((u >> 16) & 1u)) >> 16;   // round-to-nearest-even
    unsigned short s = (unsigned short)r;
    return __builtin_bit_cast(bf16_t, s);
}

// A fragment (16x32 bf16, M x K). ISA 7.12.2: lane l(0..15) holds row M=l,
// K in {0..7, 16..23}; lane l+16 holds row M=l, K in {8..15, 24..31}.
static __device__ __forceinline__ v16bf load_frag_a(const bf16_t* A, int lda, int m0, int kk) {
    int lane = threadIdx.x & 31;
    const bf16_t* p = A + (long)(m0 + (lane & 15)) * lda + kk + ((lane >> 4) << 3);
    union { v16bf v; v8bf h[2]; } u;
    u.h[0] = *(const v8bf*)p;        // K = koff .. koff+7
    u.h[1] = *(const v8bf*)(p + 16); // K = koff+16 .. koff+23
    return u.v;
}

// B fragment (32x16 bf16, K x N), sourced from row-major weight W[N,K] (B=W^T).
// lane n(0..15) holds column N=n, K in {0..7, 16..23}; lane n+16 holds N=n, K in {8..15, 24..31}.
static __device__ __forceinline__ v16bf load_frag_b(const bf16_t* W, int ldw, int n0, int kk) {
    int lane = threadIdx.x & 31;
    const bf16_t* p = W + (long)(n0 + (lane & 15)) * ldw + kk + ((lane >> 4) << 3);
    union { v16bf v; v8bf h[2]; } u;
    u.h[0] = *(const v8bf*)p;        // K = koff .. koff+7
    u.h[1] = *(const v8bf*)(p + 16); // K = koff+16 .. koff+23
    return u.v;
}

// C/D tile (16x16 f32): VGPR r, lanes 0-15 -> M=m0+r, lanes 16-31 -> M=m0+8+r.
static __device__ __forceinline__ v8f load_cin(const float* Cin, long ld, int m0, int n0) {
    int lane = threadIdx.x & 31;
    int n  = n0 + (lane & 15);
    int mb = m0 + ((lane >> 4) << 3);
    v8f r;
#pragma unroll
    for (int i = 0; i < 8; ++i) r[i] = Cin[(long)(mb + i) * ld + n];
    return r;
}

static __device__ __forceinline__ void store_tile(float* C, long ldc, int m0, int n0,
                                                  const v8f& acc, const float* bias) {
    int lane = threadIdx.x & 31;
    int n  = n0 + (lane & 15);
    int mb = m0 + ((lane >> 4) << 3);
    float bv = bias ? bias[n] : 0.f;
#pragma unroll
    for (int r = 0; r < 8; ++r) C[(long)(mb + r) * ldc + n] = acc[r] + bv;
}

// ---------------------------------------------------------------------------
// Generic bf16 WMMA GEMM: Cout[M,N] = A[M,K] @ W[N,K]^T (+Cin) (+bias[n]).
// Each wave computes a 16x64 strip (4 accumulators). The K-loop is a manual
// 2x-unrolled ping-pong: buffers {a0,b0}/{a1,b1} alternate producer/consumer
// roles so their registers are loop-invariant (no rotation movs) and each
// buffer's global_load clause overlaps the other buffer's 4 WMMAs.
// Requires K % 64 == 0 (holds for K = 512 / 1024 here).
// STAGE_A: the whole (small, M<=32) A panel is copied to LDS once per block
// with gfx1250 async-to-LDS loads (ASYNCcnt), then fragments come from
// ds_load_b128 and are reused by every N-tile in the block.
// ---------------------------------------------------------------------------
template <bool STAGE_A>
__global__ void __launch_bounds__(256)
gemm_bf16(const bf16_t* __restrict__ A, int lda, int M,
          const bf16_t* __restrict__ W, int ldw, int N, int K,
          const float* __restrict__ Cin, long ldcin,
          float* __restrict__ Cout, long ldcout,
          const float* __restrict__ bias)
{
    extern __shared__ bf16_t smA[];
    if (STAGE_A) {
        // Async copy A[M*K] (dense, lda==K) into LDS, 16B per lane per issue.
        int chunks = (M * K) >> 3;                       // 16-byte chunks
        unsigned lds_base = (unsigned)(uintptr_t)(&smA[0]); // low 32b = LDS offset
        for (int i = threadIdx.x; i < chunks; i += blockDim.x) {
            unsigned long long gaddr = (unsigned long long)(uintptr_t)(A + (long)i * 8);
            unsigned laddr = lds_base + (unsigned)i * 16u;
            asm volatile("global_load_async_to_lds_b128 %0, %1, off"
                         :: "v"(laddr), "v"(gaddr) : "memory");
        }
        asm volatile("s_wait_asynccnt 0x0" ::: "memory");
        __syncthreads();
    }

    int wave   = threadIdx.x >> 5;
    int gw     = blockIdx.x * (blockDim.x >> 5) + wave;
    int mtiles = M >> 4;
    int nblk   = N >> 6;
    if (gw >= mtiles * nblk) return;
    int m0 = (gw % mtiles) << 4;
    int n0 = (gw / mtiles) << 6;

    v8f acc[4];
    if (Cin) {
#pragma unroll
        for (int j = 0; j < 4; ++j) acc[j] = load_cin(Cin, ldcin, m0, n0 + 16 * j);
    } else {
#pragma unroll
        for (int j = 0; j < 4; ++j)
#pragma unroll
            for (int r = 0; r < 8; ++r) acc[j][r] = 0.f;
    }

    const bf16_t* Ab = STAGE_A ? (const bf16_t*)smA : A;
    int la = STAGE_A ? K : lda;

    // ---- ping-pong pipelined K loop (2 chunks of 32 per iteration) ----
    v16bf a0 = load_frag_a(Ab, la, m0, 0);
    v16bf b0[4];
#pragma unroll
    for (int j = 0; j < 4; ++j) b0[j] = load_frag_b(W, ldw, n0 + 16 * j, 0);

    v16bf a1;
    v16bf b1[4];
    int kk = 0;
    for (; kk + 64 < K; kk += 64) {
        __builtin_prefetch(W + (long)n0 * ldw + kk + 128, 0, 1);
        a1 = load_frag_a(Ab, la, m0, kk + 32);
#pragma unroll
        for (int j = 0; j < 4; ++j) b1[j] = load_frag_b(W, ldw, n0 + 16 * j, kk + 32);
#pragma unroll
        for (int j = 0; j < 4; ++j)
            acc[j] = __builtin_amdgcn_wmma_f32_16x16x32_bf16(
                false, a0, false, b0[j], (short)0, acc[j], false, false);

        a0 = load_frag_a(Ab, la, m0, kk + 64);
#pragma unroll
        for (int j = 0; j < 4; ++j) b0[j] = load_frag_b(W, ldw, n0 + 16 * j, kk + 64);
#pragma unroll
        for (int j = 0; j < 4; ++j)
            acc[j] = __builtin_amdgcn_wmma_f32_16x16x32_bf16(
                false, a1, false, b1[j], (short)0, acc[j], false, false);
    }
    // tail: chunks kk and kk+32 (K % 64 == 0)
    a1 = load_frag_a(Ab, la, m0, kk + 32);
#pragma unroll
    for (int j = 0; j < 4; ++j) b1[j] = load_frag_b(W, ldw, n0 + 16 * j, kk + 32);
#pragma unroll
    for (int j = 0; j < 4; ++j)
        acc[j] = __builtin_amdgcn_wmma_f32_16x16x32_bf16(
            false, a0, false, b0[j], (short)0, acc[j], false, false);
#pragma unroll
    for (int j = 0; j < 4; ++j)
        acc[j] = __builtin_amdgcn_wmma_f32_16x16x32_bf16(
            false, a1, false, b1[j], (short)0, acc[j], false, false);

#pragma unroll
    for (int j = 0; j < 4; ++j) store_tile(Cout, ldcout, m0, n0 + 16 * j, acc[j], bias);
}

// ---------------------------------------------------------------------------
// Small helper kernels
// ---------------------------------------------------------------------------
__global__ void k_convert(const float* __restrict__ src, bf16_t* __restrict__ dst, long n) {
    long i = (long)blockIdx.x * blockDim.x + threadIdx.x;
    long stride = (long)gridDim.x * blockDim.x;
    for (; i < n; i += stride) dst[i] = f2bf(src[i]);
}

// Wcat[r, 0:H] = Wih[r,:], Wcat[r, H:2H] = Whh[r,:]  (rows = 4H)
__global__ void k_concat_w(const float* __restrict__ Wih, const float* __restrict__ Whh,
                           bf16_t* __restrict__ Wcat, int H, long n) {
    long i = (long)blockIdx.x * blockDim.x + threadIdx.x;
    long stride = (long)gridDim.x * blockDim.x;
    int H2 = 2 * H;
    for (; i < n; i += stride) {
        long r = i / H2;
        int  k = (int)(i % H2);
        float v = (k < H) ? Wih[r * H + k] : Whh[r * H + (k - H)];
        Wcat[i] = f2bf(v);
    }
}

// x_emb[b*S+s, :] = bf16(emb[x[b,s], :])
__global__ void k_enc_embed(const int* __restrict__ x, const float* __restrict__ emb,
                            bf16_t* __restrict__ x_emb, int H, long n) {
    long i = (long)blockIdx.x * blockDim.x + threadIdx.x;
    long stride = (long)gridDim.x * blockDim.x;
    for (; i < n; i += stride) {
        long row = i / H;
        int  k   = (int)(i % H);
        x_emb[i] = f2bf(emb[(long)x[row] * H + k]);
    }
}

__global__ void k_init(float* __restrict__ h, float* __restrict__ c,
                       bf16_t* __restrict__ hbf, int* __restrict__ tok,
                       const int* __restrict__ sos, int total, int B) {
    int i = blockIdx.x * blockDim.x + threadIdx.x;
    if (i < total) { h[i] = 0.f; c[i] = 0.f; hbf[i] = f2bf(0.f); }
    if (i < B) tok[i] = sos[0];
}

// A_dec[b, 0:H] = bf16(emb[tok[b], :]) ; A_dec[b, H:2H] = h_bf16[b, :]
__global__ void k_dec_embed(const int* __restrict__ tok, const float* __restrict__ emb,
                            const bf16_t* __restrict__ hbf, bf16_t* __restrict__ A, int H) {
    int i = blockIdx.x * blockDim.x + threadIdx.x;
    int b = i / H, k = i % H;
    if (b >= 32) return;
    A[(long)b * 2 * H + k]     = f2bf(emb[(long)tok[b] * H + k]);
    A[(long)b * 2 * H + H + k] = hbf[(long)b * H + k];
}

// LSTM gate nonlinearity; gates [32,4H] preactivations (no bias yet).
__global__ void k_lstm_cell(const float* __restrict__ gates,
                            const float* __restrict__ bih, const float* __restrict__ bhh,
                            float* __restrict__ h, float* __restrict__ c,
                            bf16_t* __restrict__ hbf,
                            float* __restrict__ enc_out, long ld_enc, int H) {
    int i = blockIdx.x * blockDim.x + threadIdx.x;
    if (i >= 32 * H) return;
    int b = i / H, n = i % H;
    const float* g = gates + (long)b * 4 * H;
    float gi = g[n]         + bih[n]         + bhh[n];
    float gf = g[H + n]     + bih[H + n]     + bhh[H + n];
    float gg = g[2 * H + n] + bih[2 * H + n] + bhh[2 * H + n];
    float go = g[3 * H + n] + bih[3 * H + n] + bhh[3 * H + n];
    float si = 1.f / (1.f + __expf(-gi));
    float sf = 1.f / (1.f + __expf(-gf));
    float so = 1.f / (1.f + __expf(-go));
    float cn = sf * c[i] + si * tanhf(gg);
    float hn = so * tanhf(cn);
    c[i] = cn; h[i] = hn; hbf[i] = f2bf(hn);
    if (enc_out) enc_out[(long)b * ld_enc + n] = hn;
}

// In-place log-softmax over logits row + argmax -> next token (one block per b).
__global__ void __launch_bounds__(512)
k_logsoftmax_argmax(float* __restrict__ logits, long ld, int V,
                    int* __restrict__ tok, float* __restrict__ out1, int t, int ML) {
    int b = blockIdx.x;
    float* row = logits + (long)b * ld;
    __shared__ float s_val[512];
    __shared__ int   s_idx[512];
    int tid = threadIdx.x;

    float mx = -3.4e38f; int mi = 0x7fffffff;
    for (int v = tid; v < V; v += blockDim.x) {
        float x = row[v];
        if (x > mx || (x == mx && v < mi)) { mx = x; mi = v; }
    }
    s_val[tid] = mx; s_idx[tid] = mi;
    __syncthreads();
    for (int s = blockDim.x >> 1; s > 0; s >>= 1) {
        if (tid < s) {
            float ov = s_val[tid + s]; int oi = s_idx[tid + s];
            if (ov > s_val[tid] || (ov == s_val[tid] && oi < s_idx[tid])) {
                s_val[tid] = ov; s_idx[tid] = oi;
            }
        }
        __syncthreads();
    }
    float rmax = s_val[0]; int ridx = s_idx[0];
    __syncthreads();

    float acc = 0.f;
    for (int v = tid; v < V; v += blockDim.x) acc += __expf(row[v] - rmax);
    s_val[tid] = acc;
    __syncthreads();
    for (int s = blockDim.x >> 1; s > 0; s >>= 1) {
        if (tid < s) s_val[tid] += s_val[tid + s];
        __syncthreads();
    }
    float lse = rmax + __logf(s_val[0]);
    for (int v = tid; v < V; v += blockDim.x) row[v] = row[v] - lse;
    if (tid == 0) { tok[b] = ridx; out1[(long)b * ML + t] = (float)ridx; }
}

// ---------------------------------------------------------------------------
// Host orchestration (graph-capture safe: kernel launches only).
// ---------------------------------------------------------------------------
static inline int cdiv(long a, long b) { return (int)((a + b - 1) / b); }

extern "C" void kernel_launch(void* const* d_in, const int* in_sizes, int n_in,
                              void* d_out, int out_size, void* d_ws, size_t ws_size,
                              hipStream_t stream) {
    const int V = 32000, H = 512, B = 32, S = 64, ML = 64;
    const int G = 4 * H;          // 2048 gate width
    const long BS = (long)B * S;  // 2048

    const int*   x        = (const int*)  d_in[0];
    const float* emb      = (const float*)d_in[1];
    const float* enc_Wih  = (const float*)d_in[2];
    const float* enc_Whh  = (const float*)d_in[3];
    const float* enc_bih  = (const float*)d_in[4];
    const float* enc_bhh  = (const float*)d_in[5];
    const float* dec_Wih  = (const float*)d_in[6];
    const float* dec_Whh  = (const float*)d_in[7];
    const float* dec_bih  = (const float*)d_in[8];
    const float* dec_bhh  = (const float*)d_in[9];
    const float* W_out    = (const float*)d_in[10];
    const float* b_out    = (const float*)d_in[11];
    const int*   dec_sos  = (const int*)  d_in[12];

    float* out1    = (float*)d_out;                       // [B, ML] token ids
    float* out2    = out1 + (long)B * ML;                 // [B, ML, V] log-probs
    float* enc_out = out2 + (long)B * ML * V;             // [B, S, H]

    char* ws = (char*)d_ws;
    size_t off = 0;
    auto alloc = [&](size_t bytes) { void* p = ws + off; off += (bytes + 255) & ~(size_t)255; return p; };
    bf16_t* wout_bf  = (bf16_t*)alloc((size_t)V * H * 2);   // 32.8 MB (fits 192MB L2)
    bf16_t* eWih_bf  = (bf16_t*)alloc((size_t)G * H * 2);
    bf16_t* eWhh_bf  = (bf16_t*)alloc((size_t)G * H * 2);
    bf16_t* wcat_bf  = (bf16_t*)alloc((size_t)G * 2 * H * 2);
    bf16_t* x_emb    = (bf16_t*)alloc((size_t)BS * H * 2);
    float*  Xe       = (float*) alloc((size_t)BS * G * 4);  // precomputed x@Wih^T, all steps
    float*  gates    = (float*) alloc((size_t)B * G * 4);
    float*  hbuf     = (float*) alloc((size_t)B * H * 4);
    float*  cbuf     = (float*) alloc((size_t)B * H * 4);
    bf16_t* hbf      = (bf16_t*)alloc((size_t)B * H * 2);
    bf16_t* Adec     = (bf16_t*)alloc((size_t)B * 2 * H * 2);
    int*    tok      = (int*)   alloc((size_t)B * 4);
    (void)ws_size; (void)in_sizes; (void)n_in; (void)out_size;

    auto launch_gemm = [&](bool stage, const bf16_t* A, int lda, int M,
                           const bf16_t* Wm, int ldw, int N, int K,
                           const float* Cin, long ldcin,
                           float* Cout, long ldcout, const float* bias) {
        int waves  = (M / 16) * (N / 64);
        int blocks = (waves + 7) / 8;
        if (stage) {
            size_t sh = (size_t)M * K * sizeof(bf16_t);
            gemm_bf16<true><<<blocks, 256, sh, stream>>>(A, lda, M, Wm, ldw, N, K,
                                                         Cin, ldcin, Cout, ldcout, bias);
        } else {
            gemm_bf16<false><<<blocks, 256, 0, stream>>>(A, lda, M, Wm, ldw, N, K,
                                                         Cin, ldcin, Cout, ldcout, bias);
        }
    };

    // --- one-time conversions / embeddings --------------------------------
    k_convert<<<2048, 256, 0, stream>>>(W_out,   wout_bf, (long)V * H);
    k_convert<<<512,  256, 0, stream>>>(enc_Wih, eWih_bf, (long)G * H);
    k_convert<<<512,  256, 0, stream>>>(enc_Whh, eWhh_bf, (long)G * H);
    k_concat_w<<<1024, 256, 0, stream>>>(dec_Wih, dec_Whh, wcat_bf, H, (long)G * 2 * H);
    k_enc_embed<<<1024, 256, 0, stream>>>(x, emb, x_emb, H, BS * H);
    k_init<<<cdiv(B * H, 256), 256, 0, stream>>>(hbuf, cbuf, hbf, tok, dec_sos, B * H, B);

    // --- encoder: bulk input GEMM, then sequential h-recurrence ------------
    // Xe[BS, 4H] = x_emb @ enc_Wih^T  (all timesteps in parallel)
    launch_gemm(false, x_emb, H, (int)BS, eWih_bf, H, G, H, nullptr, 0, Xe, G, nullptr);

    for (int t = 0; t < S; ++t) {
        // gates = Xe[:, t] + h @ enc_Whh^T   (Cin = Xe slice, ld = S*4H)
        launch_gemm(true, hbf, H, B, eWhh_bf, H, G, H,
                    Xe + (long)t * G, (long)S * G, gates, G, nullptr);
        k_lstm_cell<<<cdiv(B * H, 256), 256, 0, stream>>>(
            gates, enc_bih, enc_bhh, hbuf, cbuf, hbf,
            enc_out + (long)t * H, (long)S * H, H);
    }

    // --- greedy decode loop -----------------------------------------------
    for (int t = 0; t < ML; ++t) {
        k_dec_embed<<<cdiv(B * H, 256), 256, 0, stream>>>(tok, emb, hbf, Adec, H);
        // gates = [e|h] @ [Wih|Whh]^T   (K = 2H = 1024)
        launch_gemm(true, Adec, 2 * H, B, wcat_bf, 2 * H, G, 2 * H,
                    nullptr, 0, gates, G, nullptr);
        k_lstm_cell<<<cdiv(B * H, 256), 256, 0, stream>>>(
            gates, dec_bih, dec_bhh, hbuf, cbuf, hbf, nullptr, 0, H);
        // logits -> out2 slice directly: [32, 32000] = h @ W_out^T + b_out
        launch_gemm(true, hbf, H, B, wout_bf, H, V, H,
                    nullptr, 0, out2 + (long)t * V, (long)ML * V, b_out);
        k_logsoftmax_argmax<<<B, 512, 0, stream>>>(
            out2 + (long)t * V, (long)ML * V, V, tok, out1, t, ML);
    }
}
